// CNNModel_35991825940846
// MI455X (gfx1250) — compile-verified, matrix operand order checked by reading, not run on hardware
//
#include <hip/hip_runtime.h>

// ---------------------------------------------------------------------------
// MI455X (gfx1250): NHWC f16 WMMA implicit-GEMM convs. conv2-4 stage the raw
// (unreplicated) input tile in LDS via the Tensor Data Mover when available
// (fallback: vectorized manual copy); conv1 uses im2col staging. WMMA FC,
// parallel synapse precompute, one sequential LTC block.
// ---------------------------------------------------------------------------

typedef __attribute__((ext_vector_type(16))) _Float16 v16h;
typedef __attribute__((ext_vector_type(8)))  _Float16 v8h;
typedef __attribute__((ext_vector_type(8)))  float    v8f;

#define T_FRAMES 512
#define LAT      32
#define UNITS    19
#define ODE      6

#if defined(__has_builtin)
#if __has_builtin(__builtin_amdgcn_tensor_load_to_lds) && \
    __has_builtin(__builtin_amdgcn_s_wait_tensorcnt)
#define HAVE_TDM 1
#endif
#endif
#ifndef HAVE_TDM
#define HAVE_TDM 0
#endif

__device__ __forceinline__ float softplusf(float x) {
  return (x > 20.f) ? x : log1pf(__expf(x));
}
__device__ __forceinline__ float sigmoidf(float x) {
  return 1.f / (1.f + __expf(-x));
}

__device__ __forceinline__ v16h make_frag(v8h lo, v8h hi) {
  v16h r;
#pragma unroll
  for (int e = 0; e < 8; ++e) { r[e] = lo[e]; r[e + 8] = hi[e]; }
  return r;
}
// Table-order fragment (CDNA5 ISA §7.12.2): lane needs 16B at p and at p+16.
__device__ __forceinline__ v16h load_frag(const _Float16* p) {
  return make_frag(*(const v8h*)(p), *(const v8h*)(p + 16));
}

// ---------------------------------------------------------------------------
// NCHW f32 -> NHWC f16 (input frames). src (T,C,HW) -> dst (T,HW,C).
// ---------------------------------------------------------------------------
__global__ void k_to_nhwc(const float* __restrict__ src,
                          _Float16* __restrict__ dst, int C, int HW) {
  long n = (long)T_FRAMES * C * HW;
  long i = (long)blockIdx.x * blockDim.x + threadIdx.x;
  long stride = (long)gridDim.x * blockDim.x;
  for (; i < n; i += stride) {
    long t = i / ((long)C * HW);
    long r = i - t * (long)C * HW;
    int c = (int)(r / HW), p = (int)(r % HW);
    dst[(t * HW + p) * C + c] = (_Float16)src[i];
  }
}

// Pack conv weights (COUT,CIN,KH,KW) f32 -> (MROWS, KPAD) f16 with
// k = tap*CIN_S + ci (tap = ky*KW + kx), zero padding everywhere else.
__global__ void k_pack_wconv(const float* __restrict__ w,
                             _Float16* __restrict__ wp,
                             int COUT, int CIN, int CIN_S, int TAPS,
                             int MROWS, int KPAD) {
  int idx = blockIdx.x * blockDim.x + threadIdx.x;
  if (idx >= MROWS * KPAD) return;
  int co = idx / KPAD, k = idx % KPAD;
  int tap = k / CIN_S, ci = k % CIN_S;
  _Float16 v = (_Float16)0.f;
  if (co < COUT && tap < TAPS && ci < CIN)
    v = (_Float16)w[(co * CIN + ci) * TAPS + tap];
  wp[idx] = v;
}

// Repack fc1w (32, c*576+pos) -> f16 (32, pos*64+c) to match NHWC features.
__global__ void k_pack_fc(const float* __restrict__ w,
                          _Float16* __restrict__ wp) {
  int idx = blockIdx.x * blockDim.x + threadIdx.x;
  if (idx >= 32 * 36864) return;
  int m = idx / 36864, k = idx % 36864;
  int pos = k / 64, c = k % 64;
  wp[idx] = (_Float16)w[m * 36864 + c * 576 + pos];
}

// ---------------------------------------------------------------------------
// im2col-in-LDS WMMA conv (used for conv1, CIN_S=3).
// ---------------------------------------------------------------------------
template <int CIN_S, int COUT, int COUT_S, int KH, int KW, int STRIDE, int PAD,
          int HIN, int WIN, int HOUT, int WOUT, int MTILES, int NTILE, int KPAD>
__global__ __launch_bounds__(32 * MTILES)
void conv_wmma_kernel(const _Float16* __restrict__ in,
                      const _Float16* __restrict__ wp,
                      const float* __restrict__ bias,
                      _Float16* __restrict__ out) {
  constexpr int TAPS = KH * KW;
  constexpr int KUSE = TAPS * CIN_S;
  constexpr int NP   = 16 * NTILE;
  constexpr int NPOS = HOUT * WOUT;
  constexpr int NTHREADS = 32 * MTILES;
  __shared__ __align__(16) _Float16 colbuf[NP][KPAD];

  const int t     = blockIdx.y;
  const int ptile = blockIdx.x;
  const int tid   = threadIdx.x;
  const _Float16* inT = in + (size_t)t * HIN * WIN * CIN_S;

  for (int e = tid; e < NP * TAPS; e += NTHREADS) {
    int p = e / TAPS, tap = e % TAPS;
    int pos = ptile * NP + p;
    int oy = pos / WOUT, ox = pos % WOUT;
    int ky = tap / KW, kx = tap % KW;
    int iy = oy * STRIDE - PAD + ky;
    int ix = ox * STRIDE - PAD + kx;
    bool inb = (iy >= 0 && iy < HIN && ix >= 0 && ix < WIN);
    const _Float16* src = inT + (size_t)(iy * WIN + ix) * CIN_S;
    _Float16* dst = &colbuf[p][tap * CIN_S];
#pragma unroll
    for (int c = 0; c < CIN_S; ++c) dst[c] = inb ? src[c] : (_Float16)0.f;
  }
  if constexpr (KPAD > KUSE) {
    constexpr int TAILK = KPAD - KUSE;
    for (int e = tid; e < NP * TAILK; e += NTHREADS)
      colbuf[e / TAILK][KUSE + e % TAILK] = (_Float16)0.f;
  }
  __syncthreads();

  const int wave = tid >> 5, lane = tid & 31;
  const int nl = lane & 15, half = lane >> 4;
  const _Float16* arow = wp + (size_t)(wave * 16 + nl) * KPAD;
  v8f acc[NTILE] = {};
  for (int ks = 0; ks < KPAD / 32; ++ks) {
    int base = ks * 32 + half * 8;
    v16h a = load_frag(arow + base);
#pragma unroll
    for (int n = 0; n < NTILE; ++n) {
      v16h b = load_frag(&colbuf[n * 16 + nl][base]);
      acc[n] = __builtin_amdgcn_wmma_f32_16x16x32_f16(
          false, a, false, b, (short)0, acc[n], false, false);
    }
  }

  const int mbase = wave * 16 + 8 * half;
  _Float16* outT = out + (size_t)t * NPOS * COUT_S;
  if (mbase < COUT_S) {
#pragma unroll
    for (int n = 0; n < NTILE; ++n) {
      int pos = ptile * NP + n * 16 + nl;
      v8h o;
#pragma unroll
      for (int r = 0; r < 8; ++r) {
        int m = mbase + r;
        float v = 0.f;
        if (m < COUT) { v = acc[n][r] + bias[m]; v = v > 0.f ? v : 0.f; }
        o[r] = (_Float16)v;
      }
      *(v8h*)(outT + (size_t)pos * COUT_S + mbase) = o;
    }
  }
}

// ---------------------------------------------------------------------------
// Tensor Data Mover: 2-D row-tile load into LDS (count=1, type=2, 2-byte
// elements). Descriptor per CDNA5 ISA §8.3/8.4. All words readfirstlane'd.
// ---------------------------------------------------------------------------
#if HAVE_TDM
typedef unsigned int u32x4 __attribute__((ext_vector_type(4)));
typedef int i32x8 __attribute__((ext_vector_type(8)));
typedef int i32x4 __attribute__((ext_vector_type(4)));

__device__ __forceinline__ void tdm_load_row(const void* gaddr, unsigned lds_byte,
                                             unsigned tensor_d0, unsigned tensor_d1,
                                             unsigned tile_d0,
                                             unsigned long long stride0) {
  unsigned long long ga = (unsigned long long)gaddr;
  unsigned g0w0 = 1u;                                   // count=1, user D#
  unsigned g0w1 = lds_byte;                             // lds_addr
  unsigned g0w2 = (unsigned)(ga & 0xffffffffu);         // global_addr[31:0]
  unsigned g0w3 = (unsigned)((ga >> 32) & 0x1ffffffu) | (2u << 30);  // +type=2
  unsigned g1w0 = (1u << 16);                           // data_size=1 (2B)
  unsigned g1w1 = (tensor_d0 & 0xffffu) << 16;          // tensor_dim0[15:0]
  unsigned g1w2 = (tensor_d0 >> 16) | ((tensor_d1 & 0xffffu) << 16);
  unsigned g1w3 = (tensor_d1 >> 16) | (tile_d0 << 16);  // tile_dim0
  unsigned g1w4 = 1u;                                   // tile_dim1 = 1
  unsigned g1w5 = (unsigned)(stride0 & 0xffffffffu);    // dim0 stride
  unsigned g1w6 = (unsigned)((stride0 >> 32) & 0xffffu);
  u32x4 g0;
  g0[0] = (unsigned)__builtin_amdgcn_readfirstlane((int)g0w0);
  g0[1] = (unsigned)__builtin_amdgcn_readfirstlane((int)g0w1);
  g0[2] = (unsigned)__builtin_amdgcn_readfirstlane((int)g0w2);
  g0[3] = (unsigned)__builtin_amdgcn_readfirstlane((int)g0w3);
  i32x8 g1;
  g1[0] = __builtin_amdgcn_readfirstlane((int)g1w0);
  g1[1] = __builtin_amdgcn_readfirstlane((int)g1w1);
  g1[2] = __builtin_amdgcn_readfirstlane((int)g1w2);
  g1[3] = __builtin_amdgcn_readfirstlane((int)g1w3);
  g1[4] = __builtin_amdgcn_readfirstlane((int)g1w4);
  g1[5] = __builtin_amdgcn_readfirstlane((int)g1w5);
  g1[6] = __builtin_amdgcn_readfirstlane((int)g1w6);
  g1[7] = 0;
  i32x4 z4 = {0, 0, 0, 0};
#if __clang_major__ >= 23
  i32x8 z8 = {0, 0, 0, 0, 0, 0, 0, 0};
  __builtin_amdgcn_tensor_load_to_lds(g0, g1, z4, z4, z8, 0);
#else
  __builtin_amdgcn_tensor_load_to_lds(g0, g1, z4, z4, 0);
#endif
}
#endif  // HAVE_TDM

// ---------------------------------------------------------------------------
// Raw-tile WMMA conv (conv2-4, CIN_S % 8 == 0). LDS holds the unreplicated
// input tile: KH rows x TILEW cols x CIN_S ch (NHWC rows contiguous). Each
// WMMA K-chunk of 8 stays inside one tap, so B fragments are two aligned
// ds_load_b128 at decomposed (tap,ci) offsets. Interior rows arrive via TDM
// (wave 0 issues one tensor_load per row, waits TENSORcnt, then barrier);
// halos/OOB/slop are zero-filled by all threads in disjoint LDS regions.
// Requires NP | WOUT so a block's positions share one output row.
// ---------------------------------------------------------------------------
template <int CIN_S, int COUT, int COUT_S, int KH, int KW, int STRIDE, int PAD,
          int HIN, int WIN, int HOUT, int WOUT, int MTILES, int NTILE, int KPAD>
__global__ __launch_bounds__(32 * MTILES)
void conv_tile_wmma(const _Float16* __restrict__ in,   // (T,HIN*WIN,CIN_S)
                    const _Float16* __restrict__ wp,   // (MTILES*16, KPAD)
                    const float* __restrict__ bias,
                    _Float16* __restrict__ out) {      // (T,NPOS,COUT_S)
  constexpr int NP    = 16 * NTILE;
  constexpr int TILEW = (NP - 1) * STRIDE + KW;
  constexpr int NPOS  = HOUT * WOUT;
  constexpr int ROWE  = TILEW * CIN_S;       // elems per tile row
  constexpr int TILEE = KH * ROWE;           // tile proper
  constexpr int SLOP  = ROWE + 64;           // zero slack for K-tail reads
  constexpr int NTHREADS = 32 * MTILES;
  static_assert(WOUT % NP == 0, "block positions must share one output row");
  static_assert(CIN_S % 8 == 0, "K-chunks must stay inside one tap");
  __shared__ __align__(16) _Float16 tile[TILEE + SLOP];  // sole LDS object

  const int t     = blockIdx.y;
  const int ptile = blockIdx.x;
  const int tid   = threadIdx.x;
  const _Float16* inT = in + (size_t)t * HIN * WIN * CIN_S;

  const int oy  = (ptile * NP) / WOUT;
  const int ox0 = (ptile * NP) % WOUT;
  const int iy0 = oy * STRIDE - PAD;
  const int ix0 = ox0 * STRIDE - PAD;
  const int xlo = ix0 < 0 ? 0 : ix0;               // in-bounds col range
  const int xhi = (ix0 + TILEW) > WIN ? WIN : (ix0 + TILEW);

  // ---- zero slack ----
  for (int c = tid; c < SLOP / 8; c += NTHREADS)
    *(v8h*)(tile + TILEE + c * 8) = v8h{};

#if HAVE_TDM
  // ---- zero halo columns and out-of-bounds rows ----
  for (int e = tid; e < KH * TILEW; e += NTHREADS) {
    int ky = e / TILEW, xo = e % TILEW;
    int iy = iy0 + ky, ix = ix0 + xo;
    if (iy < 0 || iy >= HIN || ix < xlo || ix >= xhi) {
#pragma unroll
      for (int c8 = 0; c8 < CIN_S / 8; ++c8)
        *(v8h*)(tile + (ky * TILEW + xo) * CIN_S + c8 * 8) = v8h{};
    }
  }
  // ---- TDM: one row-tile DMA per in-bounds row; wave 0 owns TENSORcnt ----
  if (tid < 32) {
    for (int ky = 0; ky < KH; ++ky) {
      int iy = iy0 + ky;
      if (iy >= 0 && iy < HIN) {
        tdm_load_row(inT + ((size_t)iy * WIN + xlo) * CIN_S,
                     (unsigned)((ky * TILEW + (xlo - ix0)) * CIN_S * 2),
                     (unsigned)(WIN * CIN_S), (unsigned)HIN,
                     (unsigned)((xhi - xlo) * CIN_S),
                     (unsigned long long)(WIN * CIN_S));
      }
    }
    __builtin_amdgcn_s_wait_tensorcnt(0);
  }
#else
  // ---- manual staging: copy/zero one (row, col) cell of CIN_S elems ----
  for (int e = tid; e < KH * TILEW; e += NTHREADS) {
    int ky = e / TILEW, xo = e % TILEW;
    int iy = iy0 + ky, ix = ix0 + xo;
    bool inb = (iy >= 0 && iy < HIN && ix >= 0 && ix < WIN);
    const _Float16* src = inT + (size_t)(iy * WIN + ix) * CIN_S;
    _Float16* dst = tile + (ky * TILEW + xo) * CIN_S;
#pragma unroll
    for (int c8 = 0; c8 < CIN_S / 8; ++c8) {
      v8h v = {};
      if (inb) v = *(const v8h*)(src + c8 * 8);
      *(v8h*)(dst + c8 * 8) = v;
    }
  }
#endif
  __syncthreads();

  const int wave = tid >> 5, lane = tid & 31;
  const int nl = lane & 15, half = lane >> 4;
  const _Float16* arow = wp + (size_t)(wave * 16 + nl) * KPAD;
  v8f acc[NTILE] = {};
  for (int ks = 0; ks < KPAD / 32; ++ks) {
    int k0 = ks * 32 + half * 8;        // elements 0..7 of fragment
    int k1 = k0 + 16;                   // elements 8..15
    v16h a = load_frag(arow + k0);
    int tap0 = k0 / CIN_S, ci0 = k0 - tap0 * CIN_S;
    int tap1 = k1 / CIN_S, ci1 = k1 - tap1 * CIN_S;
    int ky0 = tap0 / KW, kx0 = tap0 - ky0 * KW;
    int ky1 = tap1 / KW, kx1 = tap1 - ky1 * KW;
    int o0 = (ky0 * TILEW + kx0) * CIN_S + ci0;
    int o1 = (ky1 * TILEW + kx1) * CIN_S + ci1;
#pragma unroll
    for (int n = 0; n < NTILE; ++n) {
      int xb = (n * 16 + nl) * STRIDE * CIN_S;
      v16h b = make_frag(*(const v8h*)(tile + o0 + xb),
                         *(const v8h*)(tile + o1 + xb));
      acc[n] = __builtin_amdgcn_wmma_f32_16x16x32_f16(
          false, a, false, b, (short)0, acc[n], false, false);
    }
  }

  const int mbase = wave * 16 + 8 * half;
  _Float16* outT = out + (size_t)t * NPOS * COUT_S;
  if (mbase < COUT_S) {
#pragma unroll
    for (int n = 0; n < NTILE; ++n) {
      int pos = ptile * NP + n * 16 + nl;
      v8h o;
#pragma unroll
      for (int r = 0; r < 8; ++r) {
        int m = mbase + r;
        float v = 0.f;
        if (m < COUT) { v = acc[n][r] + bias[m]; v = v > 0.f ? v : 0.f; }
        o[r] = (_Float16)v;
      }
      *(v8h*)(outT + (size_t)pos * COUT_S + mbase) = o;   // global_store_b128
    }
  }
}

// ---------------------------------------------------------------------------
// FC + input affine. M=32 (LAT), N=512 frames, K=36864 (NHWC flatten).
// ---------------------------------------------------------------------------
__global__ __launch_bounds__(64)
void k_fc_wmma(const _Float16* __restrict__ feat,
               const _Float16* __restrict__ wp,
               const float* __restrict__ fc1b,
               const float* __restrict__ in_w, const float* __restrict__ in_b,
               float* __restrict__ useq) {
  constexpr int KTOT = 576 * 64;
  const int ntile = blockIdx.x;
  const int wave  = threadIdx.x >> 5;
  const int lane  = threadIdx.x & 31;
  const int nl = lane & 15, half = lane >> 4;

  const _Float16* arow = wp   + (size_t)(wave * 16 + nl) * KTOT;
  const _Float16* brow = feat + (size_t)(ntile * 16 + nl) * KTOT;

  v8f acc = {};
#pragma unroll 4
  for (int ks = 0; ks < KTOT / 32; ++ks) {
    int base = ks * 32 + half * 8;
    __builtin_prefetch(brow + base + 512, 0, 1);  // global_prefetch_b8
    v16h a = load_frag(arow + base);
    v16h b = load_frag(brow + base);
    acc = __builtin_amdgcn_wmma_f32_16x16x32_f16(
        false, a, false, b, (short)0, acc, false, false);
  }

  int t = ntile * 16 + nl;
#pragma unroll
  for (int r = 0; r < 8; ++r) {
    int m = wave * 16 + r + 8 * half;
    float f = acc[r] + fc1b[m];
    useq[t * LAT + m] = f * in_w[m] + in_b[m];
  }
}

// ---------------------------------------------------------------------------
// Sensory-synapse precompute: depends only on u_t, fully parallel over t.
// ---------------------------------------------------------------------------
__global__ __launch_bounds__(32)
void k_syn(const float* __restrict__ useq,
           const float* __restrict__ s_mu, const float* __restrict__ s_sigma,
           const float* __restrict__ s_w, const float* __restrict__ s_erev,
           const float* __restrict__ smask,
           float* __restrict__ wnum, float* __restrict__ wden) {
  int t = blockIdx.x, j = threadIdx.x;
  if (j >= UNITS) return;
  float num = 0.f, den = 0.f;
  for (int i = 0; i < LAT; ++i) {
    int ij = i * UNITS + j;
    float swp = softplusf(s_w[ij]) * smask[ij];
    float sa  = swp * sigmoidf(s_sigma[ij] * (useq[t * LAT + i] - s_mu[ij]));
    num += sa * s_erev[ij];
    den += sa;
  }
  wnum[t * UNITS + j] = num;
  wden[t * UNITS + j] = den;
}

// ---------------------------------------------------------------------------
// Sequential LTC solve: one persistent block, one thread per (pre,post) pair,
// ds_add_f32 LDS reductions, 512 steps x 6 unfolds.
// ---------------------------------------------------------------------------
__global__ __launch_bounds__(384)
void k_ltc(const float* __restrict__ wnum, const float* __restrict__ wden,
           const float* __restrict__ mu, const float* __restrict__ sigma,
           const float* __restrict__ w_rec, const float* __restrict__ erev,
           const float* __restrict__ mask,
           const float* __restrict__ gleak, const float* __restrict__ vleak,
           const float* __restrict__ cm,
           const float* __restrict__ out_w, const float* __restrict__ out_b,
           float* __restrict__ out) {
  __shared__ float v_sh[UNITS], num_sh[UNITS], den_sh[UNITS];
  __shared__ float cmt_sh[UNITS], gl_sh[UNITS], glvl_sh[UNITS];
  const int tid = threadIdx.x;
  const int i = tid / UNITS, j = tid % UNITS;
  const bool active = tid < UNITS * UNITS;

  float wp = 0.f, sg = 0.f, mu_ = 0.f, er = 0.f;
  if (active) {
    int ij = i * UNITS + j;
    wp = softplusf(w_rec[ij]) * mask[ij];
    sg = sigma[ij]; mu_ = mu[ij]; er = erev[ij];
  }
  if (tid < UNITS) {
    float c = softplusf(cm[tid]) * (float)ODE;
    float g = softplusf(gleak[tid]);
    cmt_sh[tid] = c; gl_sh[tid] = g; glvl_sh[tid] = g * vleak[tid];
    v_sh[tid] = 0.f;
  }
  __syncthreads();

  for (int t = 0; t < T_FRAMES; ++t) {
#pragma unroll 1
    for (int u = 0; u < ODE; ++u) {
      if (tid < UNITS) {
        num_sh[tid] = cmt_sh[tid] * v_sh[tid] + glvl_sh[tid] + wnum[t * UNITS + tid];
        den_sh[tid] = cmt_sh[tid] + gl_sh[tid] + wden[t * UNITS + tid] + 1e-8f;
      }
      __syncthreads();
      if (active) {
        float wa = wp * sigmoidf(sg * (v_sh[i] - mu_));
        atomicAdd(&num_sh[j], wa * er);
        atomicAdd(&den_sh[j], wa);
      }
      __syncthreads();
      if (tid < UNITS) v_sh[tid] = num_sh[tid] / den_sh[tid];
      __syncthreads();
    }
    if (tid < 2) out[t * 2 + tid] = v_sh[tid] * out_w[tid] + out_b[tid];
  }
}

// ---------------------------------------------------------------------------
// Orchestration
// ---------------------------------------------------------------------------
extern "C" void kernel_launch(void* const* d_in, const int* in_sizes, int n_in,
                              void* d_out, int out_size, void* d_ws, size_t ws_size,
                              hipStream_t stream) {
  (void)in_sizes; (void)n_in; (void)out_size; (void)ws_size;
  const float* x       = (const float*)d_in[0];
  const float* c1w     = (const float*)d_in[1];
  const float* c1b     = (const float*)d_in[2];
  const float* c2w     = (const float*)d_in[3];
  const float* c2b     = (const float*)d_in[4];
  const float* c3w     = (const float*)d_in[5];
  const float* c3b     = (const float*)d_in[6];
  const float* c4w     = (const float*)d_in[7];
  const float* c4b     = (const float*)d_in[8];
  const float* fc1w    = (const float*)d_in[9];
  const float* fc1b    = (const float*)d_in[10];
  const float* input_w = (const float*)d_in[11];
  const float* input_b = (const float*)d_in[12];
  const float* s_mu    = (const float*)d_in[13];
  const float* s_sigma = (const float*)d_in[14];
  const float* s_w     = (const float*)d_in[15];
  const float* s_erev  = (const float*)d_in[16];
  const float* mu      = (const float*)d_in[17];
  const float* sigma   = (const float*)d_in[18];
  const float* w_rec   = (const float*)d_in[19];
  const float* erev    = (const float*)d_in[20];
  const float* gleak   = (const float*)d_in[21];
  const float* vleak   = (const float*)d_in[22];
  const float* cm      = (const float*)d_in[23];
  const float* out_w   = (const float*)d_in[24];
  const float* out_b   = (const float*)d_in[25];
  const float* smask   = (const float*)d_in[26];
  const float* mask    = (const float*)d_in[27];
  float* out = (float*)d_out;

  char* ws = (char*)d_ws;
  size_t off = 0;
  auto wsalloc = [&](size_t bytes) -> char* {
    char* p = ws + off;
    off += (bytes + 255) & ~(size_t)255;
    return p;
  };

  // NHWC activations; channel dims padded to multiples of 8 (zero-filled).
  _Float16* xh   = (_Float16*)wsalloc((size_t)T_FRAMES * 36864 * 3  * 2);
  _Float16* act1 = (_Float16*)wsalloc((size_t)T_FRAMES * 9216  * 24 * 2);
  _Float16* act2 = (_Float16*)wsalloc((size_t)T_FRAMES * 2304  * 40 * 2);
  _Float16* act3 = (_Float16*)wsalloc((size_t)T_FRAMES * 576   * 48 * 2);
  _Float16* act4 = (_Float16*)wsalloc((size_t)T_FRAMES * 576   * 64 * 2);
  _Float16* w1p  = (_Float16*)wsalloc((size_t)32 * 96   * 2);
  _Float16* w2p  = (_Float16*)wsalloc((size_t)48 * 608  * 2);
  _Float16* w3p  = (_Float16*)wsalloc((size_t)48 * 1024 * 2);
  _Float16* w4p  = (_Float16*)wsalloc((size_t)64 * 448  * 2);
  _Float16* fcp  = (_Float16*)wsalloc((size_t)32 * 36864 * 2);
  float* useq = (float*)wsalloc((size_t)T_FRAMES * LAT * 4);
  float* wnum = (float*)wsalloc((size_t)T_FRAMES * UNITS * 4);
  float* wden = (float*)wsalloc((size_t)T_FRAMES * UNITS * 4);

  // ---- layout conversion + weight packing ----
  k_to_nhwc<<<32768, 256, 0, stream>>>(x, xh, 3, 144 * 256);
  k_pack_fc<<<(32 * 36864 + 255) / 256, 256, 0, stream>>>(fc1w, fcp);
  k_pack_wconv<<<(32 * 96   + 255) / 256, 256, 0, stream>>>(c1w, w1p, 24, 3,  3,  25, 32, 96);
  k_pack_wconv<<<(48 * 608  + 255) / 256, 256, 0, stream>>>(c2w, w2p, 36, 24, 24, 25, 48, 608);
  k_pack_wconv<<<(48 * 1024 + 255) / 256, 256, 0, stream>>>(c3w, w3p, 48, 36, 40, 25, 48, 1024);
  k_pack_wconv<<<(64 * 448  + 255) / 256, 256, 0, stream>>>(c4w, w4p, 64, 48, 48, 9,  64, 448);

  // ---- WMMA conv stack (NHWC) ----
  conv_wmma_kernel<3, 24, 24, 5, 5, 2, 2, 144, 256, 72, 128, 2, 2, 96>
      <<<dim3(288, T_FRAMES), 64, 0, stream>>>(xh, w1p, c1b, act1);
  conv_tile_wmma<24, 36, 40, 5, 5, 2, 2, 72, 128, 36, 64, 3, 2, 608>
      <<<dim3(72, T_FRAMES), 96, 0, stream>>>(act1, w2p, c2b, act2);
  conv_tile_wmma<40, 48, 48, 5, 5, 2, 2, 36, 64, 18, 32, 3, 2, 1024>
      <<<dim3(18, T_FRAMES), 96, 0, stream>>>(act2, w3p, c3b, act3);
  conv_tile_wmma<48, 64, 64, 3, 3, 1, 1, 18, 32, 18, 32, 4, 2, 448>
      <<<dim3(18, T_FRAMES), 128, 0, stream>>>(act3, w4p, c4b, act4);

  // ---- WMMA FC + input affine ----
  k_fc_wmma<<<32, 64, 0, stream>>>(act4, fcp, fc1b, input_w, input_b, useq);

  // ---- LTC: parallel synapse terms, then sequential solve ----
  k_syn<<<T_FRAMES, 32, 0, stream>>>(useq, s_mu, s_sigma, s_w, s_erev, smask,
                                     wnum, wden);
  k_ltc<<<1, 384, 0, stream>>>(wnum, wden, mu, sigma, w_rec, erev, mask,
                               gleak, vleak, cm, out_w, out_b, out);
}
